// MultiHeadAttention_21801253994998
// MI455X (gfx1250) — matrix operand order repaired and checked by reading, not compiled, not run on hardware
//
#include <hip/hip_runtime.h>
#include <hip/hip_bf16.h>

// ---------------------------------------------------------------------------
// MHA for gfx1250 (CDNA5): f16 WMMA GEMMs + single-wave flash attention.
// D_MODEL=1024, H=16, d=64, B=2, S=2048 (M = B*S = 4096).
// ---------------------------------------------------------------------------

typedef __attribute__((ext_vector_type(16))) _Float16 v16h;
typedef __attribute__((ext_vector_type(8)))  _Float16 v8h;
typedef __attribute__((ext_vector_type(4)))  _Float16 v4h;
typedef __attribute__((ext_vector_type(8)))  float    v8f;

#define S_LEN   2048
#define D_MODEL 1024
#define NHEADS  16
#define HDIM    64
#define M_ROWS  4096   // B * S

__device__ __forceinline__ v8f wmma_f16(v16h a, v16h b, v8f c) {
  // (neg_a, A, neg_b, B, c_mod, C, reuse_a, reuse_b)
  return __builtin_amdgcn_wmma_f32_16x16x32_f16(false, a, false, b, (short)0, c,
                                                false, false);
}

// A-fragment (16x32 f16, M x K), row-major source with row stride ldk.
// Layout: lane m (0-15)/m+16 holds row m; halves[0..15] = K = g*16..g*16+15
// (g = lane>>4), matching load_b_frag.
__device__ __forceinline__ v16h load_a_frag(const _Float16* __restrict__ base,
                                            int ldk, int lane) {
  int m = lane & 15, g = lane >> 4;
  const _Float16* p = base + (size_t)m * ldk + g * 16;
  union { v16h v; v8h h[2]; } u;
  u.h[0] = *(const v8h*)p;
  u.h[1] = *(const v8h*)(p + 8);
  return u.v;
}

// B-fragment (32x16 f16, K x N). Source holds column n of B as a contiguous
// row (i.e. B = RowsOfSource^T): lane n loads 16 contiguous K values at
// K offset g*16. base points at (n0, k0) of the source matrix.
__device__ __forceinline__ v16h load_b_frag(const _Float16* __restrict__ base,
                                            int ldk, int lane) {
  int n = lane & 15, g = lane >> 4;
  return *(const v16h*)(base + (size_t)n * ldk + g * 16);
}

// XOR-butterfly reductions over the 16-lane half-wave via ds_swizzle_b32
// (group-of-32 mode, and_mask=0x1f; xor masks < 16 never cross the half-wave).
template <int PAT>
__device__ __forceinline__ float swz(float x) {
  return __int_as_float(__builtin_amdgcn_ds_swizzle(__float_as_int(x), PAT));
}
__device__ __forceinline__ float rmax16(float v) {
  v = fmaxf(v, swz<0x201f>(v));   // xor 8
  v = fmaxf(v, swz<0x101f>(v));   // xor 4
  v = fmaxf(v, swz<0x081f>(v));   // xor 2
  v = fmaxf(v, swz<0x041f>(v));   // xor 1
  return v;
}
__device__ __forceinline__ float rsum16(float v) {
  v += swz<0x201f>(v);
  v += swz<0x101f>(v);
  v += swz<0x081f>(v);
  v += swz<0x041f>(v);
  return v;
}

// ---------------------------------------------------------------------------
// fp32 -> f16 cast, 4 elements per thread (all element counts are /4).
// ---------------------------------------------------------------------------
__global__ void cast_f32_f16_kernel(const float* __restrict__ in,
                                    _Float16* __restrict__ out, int n4) {
  int i = blockIdx.x * blockDim.x + threadIdx.x;
  if (i < n4) {
    float4 f = ((const float4*)in)[i];
    v4h o;
    o[0] = (_Float16)f.x; o[1] = (_Float16)f.y;
    o[2] = (_Float16)f.z; o[3] = (_Float16)f.w;
    ((v4h*)out)[i] = o;
  }
}

// ---------------------------------------------------------------------------
// Y = X(f16)[4096,1024] @ W(f16,[N,K] row-major)^T + bias(f32)
// One wave -> 16x64 output tile. OUT_MODE: 0 = f32 row-major (final output),
// 1 = f16 row-major (Q,K), 2 = f16 transposed per-head [B,H,d,S] (V^T).
// ---------------------------------------------------------------------------
template <int OUT_MODE>
__global__ __launch_bounds__(256) void gemm_xwT_kernel(
    const _Float16* __restrict__ X, const _Float16* __restrict__ W,
    const float* __restrict__ bias, void* __restrict__ Y) {
  int lane  = threadIdx.x & 31;
  int wid   = (blockIdx.x * 256 + threadIdx.x) >> 5;  // 4096 waves
  int mTile = (wid & 255) * 16;                        // 256 M tiles
  int nQuad = (wid >> 8) * 64;                         // 16 N quads

  v8f acc[4] = {};
  for (int k0 = 0; k0 < D_MODEL; k0 += 32) {
    v16h a = load_a_frag(X + (size_t)mTile * D_MODEL + k0, D_MODEL, lane);
#pragma unroll
    for (int t = 0; t < 4; ++t) {
      v16h b = load_b_frag(W + (size_t)(nQuad + t * 16) * D_MODEL + k0,
                           D_MODEL, lane);
      acc[t] = wmma_f16(a, b, acc[t]);
    }
  }

  int n = lane & 15, g = lane >> 4;
#pragma unroll
  for (int t = 0; t < 4; ++t) {
    int col = nQuad + t * 16 + n;
    float bv = bias[col];
#pragma unroll
    for (int r = 0; r < 8; ++r) {
      int row = mTile + r + 8 * g;   // C layout: lanes 16-31 hold rows +8
      float val = acc[t][r] + bv;
      if (OUT_MODE == 0) {
        ((float*)Y)[(size_t)row * D_MODEL + col] = val;
      } else if (OUT_MODE == 1) {
        ((_Float16*)Y)[(size_t)row * D_MODEL + col] = (_Float16)val;
      } else {
        int bb = row >> 11, s = row & (S_LEN - 1);
        int hh = col >> 6,  dd = col & (HDIM - 1);
        ((_Float16*)Y)[(((size_t)(bb * NHEADS + hh)) * HDIM + dd) * S_LEN + s] =
            (_Float16)val;
      }
    }
  }
}

// ---------------------------------------------------------------------------
// Flash attention: one wave per (b, h, 16-query tile). Streams keys in blocks
// of 64: 8 score WMMAs + one online-softmax update + LDS re-layout of P +
// 8 P*V WMMAs. Q is prescaled by (1/sqrt(d))*log2(e) so scores are already in
// the log2 domain and every exponential is a single v_exp_f32 (exp2).
// Q,K: f16 [B,S,H*d] row-major. VT: f16 [B,H,d,S]. O: f16 [B,S,H*d].
// ---------------------------------------------------------------------------
__global__ __launch_bounds__(32) void flash_attn_kernel(
    const _Float16* __restrict__ Q, const _Float16* __restrict__ K,
    const _Float16* __restrict__ VT, _Float16* __restrict__ O) {
  __shared__ __align__(32) _Float16 ldsP[16 * 64];   // P tile 16x64

  int lane = threadIdx.x & 31;
  int n = lane & 15, g = lane >> 4;

  int bhq = blockIdx.x;                  // 2*16*128 = 4096
  int q0 = (bhq & 127) * 16;
  int h  = (bhq >> 7) & 15;
  int b  = bhq >> 11;

  const _Float16* Qb = Q + ((size_t)(b * S_LEN + q0)) * D_MODEL + h * HDIM;
  const _Float16* Kb = K + ((size_t)b * S_LEN) * D_MODEL + h * HDIM;
  const _Float16* Vb = VT + ((size_t)(b * NHEADS + h)) * HDIM * S_LEN;

  // Q tile 16x64 as two A-fragments, prescaled by 0.125 * log2(e).
  const _Float16 qs = (_Float16)0.18033688f;
  v16h aQ0 = load_a_frag(Qb + 0,  D_MODEL, lane) * qs;
  v16h aQ1 = load_a_frag(Qb + 32, D_MODEL, lane) * qs;

  v8f o0 = {}, o1 = {}, o2 = {}, o3 = {};
  float mrow[8], lrow[8];
#pragma unroll
  for (int r = 0; r < 8; ++r) { mrow[r] = -1e30f; lrow[r] = 0.0f; }

  for (int kb = 0; kb < S_LEN; kb += 64) {
    // ---- scores (log2 domain): 16 q-rows x 64 keys = four 16x16 C tiles
    v8f c0 = {}, c1 = {}, c2 = {}, c3 = {};
    {
      const _Float16* kp = Kb + (size_t)kb * D_MODEL;
      c0 = wmma_f16(aQ0, load_b_frag(kp + 0 * D_MODEL + 0,  D_MODEL, lane), c0);
      c0 = wmma_f16(aQ1, load_b_frag(kp + 0 * D_MODEL + 32, D_MODEL, lane), c0);
      c1 = wmma_f16(aQ0, load_b_frag(kp + 16 * D_MODEL + 0,  D_MODEL, lane), c1);
      c1 = wmma_f16(aQ1, load_b_frag(kp + 16 * D_MODEL + 32, D_MODEL, lane), c1);
      c2 = wmma_f16(aQ0, load_b_frag(kp + 32 * D_MODEL + 0,  D_MODEL, lane), c2);
      c2 = wmma_f16(aQ1, load_b_frag(kp + 32 * D_MODEL + 32, D_MODEL, lane), c2);
      c3 = wmma_f16(aQ0, load_b_frag(kp + 48 * D_MODEL + 0,  D_MODEL, lane), c3);
      c3 = wmma_f16(aQ1, load_b_frag(kp + 48 * D_MODEL + 32, D_MODEL, lane), c3);
    }

    // ---- online softmax (base 2). Row r+8g lives in 16 lanes of half-wave g.
#pragma unroll
    for (int r = 0; r < 8; ++r) {
      float s0 = c0[r], s1 = c1[r], s2 = c2[r], s3 = c3[r];
      float mloc = rmax16(fmaxf(fmaxf(s0, s1), fmaxf(s2, s3)));
      float mnew = fmaxf(mrow[r], mloc);
      float alpha = exp2f(mrow[r] - mnew);
      float p0 = exp2f(s0 - mnew);
      float p1 = exp2f(s1 - mnew);
      float p2 = exp2f(s2 - mnew);
      float p3 = exp2f(s3 - mnew);
      float rs = rsum16((p0 + p1) + (p2 + p3));
      lrow[r] = lrow[r] * alpha + rs;
      mrow[r] = mnew;
      o0[r] *= alpha; o1[r] *= alpha; o2[r] *= alpha; o3[r] *= alpha;
      int row = r + 8 * g;
      ldsP[row * 64 + 0  + n] = (_Float16)p0;   // cols kb+ 0..15
      ldsP[row * 64 + 16 + n] = (_Float16)p1;   // cols kb+16..31
      ldsP[row * 64 + 32 + n] = (_Float16)p2;   // cols kb+32..47
      ldsP[row * 64 + 48 + n] = (_Float16)p3;   // cols kb+48..63
    }
    __syncthreads();

    // ---- re-read P (16x64) as two A-fragments
    v16h aP0, aP1;
    {
      const _Float16* lp = &ldsP[(size_t)(lane & 15) * 64 + g * 16];
      union { v16h v; v8h h2[2]; } u;
      u.h2[0] = *(const v8h*)lp;
      u.h2[1] = *(const v8h*)(lp + 8);
      aP0 = u.v;
      u.h2[0] = *(const v8h*)(lp + 32);
      u.h2[1] = *(const v8h*)(lp + 40);
      aP1 = u.v;
    }

    // ---- O += P * V  (V^T rows contiguous over s)
    o0 = wmma_f16(aP0, load_b_frag(Vb + (size_t)0  * S_LEN + kb,      S_LEN, lane), o0);
    o0 = wmma_f16(aP1, load_b_frag(Vb + (size_t)0  * S_LEN + kb + 32, S_LEN, lane), o0);
    o1 = wmma_f16(aP0, load_b_frag(Vb + (size_t)16 * S_LEN + kb,      S_LEN, lane), o1);
    o1 = wmma_f16(aP1, load_b_frag(Vb + (size_t)16 * S_LEN + kb + 32, S_LEN, lane), o1);
    o2 = wmma_f16(aP0, load_b_frag(Vb + (size_t)32 * S_LEN + kb,      S_LEN, lane), o2);
    o2 = wmma_f16(aP1, load_b_frag(Vb + (size_t)32 * S_LEN + kb + 32, S_LEN, lane), o2);
    o3 = wmma_f16(aP0, load_b_frag(Vb + (size_t)48 * S_LEN + kb,      S_LEN, lane), o3);
    o3 = wmma_f16(aP1, load_b_frag(Vb + (size_t)48 * S_LEN + kb + 32, S_LEN, lane), o3);
    __syncthreads();
  }

  // ---- epilogue: normalize and store f16 [B,S,H*d]
#pragma unroll
  for (int r = 0; r < 8; ++r) {
    float inv = 1.0f / lrow[r];
    int row = q0 + r + 8 * g;
    size_t base = ((size_t)(b * S_LEN + row)) * D_MODEL + h * HDIM + n;
    O[base + 0]  = (_Float16)(o0[r] * inv);
    O[base + 16] = (_Float16)(o1[r] * inv);
    O[base + 32] = (_Float16)(o2[r] * inv);
    O[base + 48] = (_Float16)(o3[r] * inv);
  }
}

// ---------------------------------------------------------------------------
extern "C" void kernel_launch(void* const* d_in, const int* in_sizes, int n_in,
                              void* d_out, int out_size, void* d_ws, size_t ws_size,
                              hipStream_t stream) {
  const float* q  = (const float*)d_in[0];
  const float* k  = (const float*)d_in[1];
  const float* v  = (const float*)d_in[2];
  const float* Wq = (const float*)d_in[3];
  const float* bq = (const float*)d_in[4];
  const float* Wk = (const float*)d_in[5];
  const float* bk = (const float*)d_in[6];
  const float* Wv = (const float*)d_in[7];
  const float* bv = (const float*)d_in[8];
  const float* Wo = (const float*)d_in[9];
  const float* bo = (const float*)d_in[10];

  char* ws = (char*)d_ws;
  const size_t MB = (size_t)1 << 20;
  // Sequential (stream-ordered) buffer reuse: ~34 MB total.
  _Float16* Xbuf = (_Float16*)(ws + 0);         // 8 MB: casted activation / attn out
  _Float16* Wbuf = (_Float16*)(ws + 8 * MB);    // 2 MB: casted weight
  _Float16* Qh   = (_Float16*)(ws + 10 * MB);   // 8 MB
  _Float16* Kh   = (_Float16*)(ws + 18 * MB);   // 8 MB
  _Float16* VTh  = (_Float16*)(ws + 26 * MB);   // 8 MB

  const int NX = M_ROWS * D_MODEL;     // 4M activation elements
  const int NW = D_MODEL * D_MODEL;    // 1M weight elements
  auto cast = [&](const float* src, _Float16* dst, int nelem) {
    int n4 = nelem / 4;
    cast_f32_f16_kernel<<<(n4 + 255) / 256, 256, 0, stream>>>(src, dst, n4);
  };

  dim3 gG(512), bG(256);  // 4096 waves = 256 Mtiles x 16 Nquads

  // Q = query @ Wq^T + bq
  cast(q, Xbuf, NX); cast(Wq, Wbuf, NW);
  gemm_xwT_kernel<1><<<gG, bG, 0, stream>>>(Xbuf, Wbuf, bq, Qh);
  // K = key @ Wk^T + bk
  cast(k, Xbuf, NX); cast(Wk, Wbuf, NW);
  gemm_xwT_kernel<1><<<gG, bG, 0, stream>>>(Xbuf, Wbuf, bk, Kh);
  // V^T = (value @ Wv^T + bv) transposed per head -> [B,H,d,S]
  cast(v, Xbuf, NX); cast(Wv, Wbuf, NW);
  gemm_xwT_kernel<2><<<gG, bG, 0, stream>>>(Xbuf, Wbuf, bv, VTh);
  // attention -> Xbuf (f16 [B,S,H*d])
  flash_attn_kernel<<<4096, 32, 0, stream>>>(Qh, Kh, VTh, Xbuf);
  // out = attn @ Wo^T + bo (f32)
  cast(Wo, Wbuf, NW);
  gemm_xwT_kernel<0><<<gG, bG, 0, stream>>>(Xbuf, Wbuf, bo, d_out);
}
